// GrokOmega_35880156790979
// MI455X (gfx1250) — matrix operand order repaired and checked by reading, not compiled, not run on hardware
//
#include <hip/hip_runtime.h>
#include <cstdint>
#include <cstddef>

typedef __attribute__((ext_vector_type(16))) __bf16 v16bf;
typedef __attribute__((ext_vector_type(8)))  float  v8f;
typedef __attribute__((ext_vector_type(4)))  unsigned int v4u;
typedef __attribute__((ext_vector_type(8)))  int  v8i;
typedef __attribute__((ext_vector_type(4)))  int  v4i;

#define T_LEN 4096
#define HID   512
#define DIM   128
#define VOC   32000
#define GRU_NBLK 16

#if defined(__has_builtin)
#if __has_builtin(__builtin_amdgcn_tensor_load_to_lds)
#define USE_TDM 1
#endif
#endif

// ---------------- cooperative grid barrier (16 blocks, poll with s_sleep) ----
__device__ inline void grid_barrier(int* cnt, int* gen) {
  __syncthreads();
  if (threadIdx.x == 0) {
    __threadfence();
    int g = atomicAdd(gen, 0);
    if (atomicAdd(cnt, 1) == GRU_NBLK - 1) {
      atomicExch(cnt, 0);
      __threadfence();
      atomicAdd(gen, 1);
    } else {
      while (atomicAdd(gen, 0) == g) { __builtin_amdgcn_s_sleep(2); }
    }
    __threadfence();
  }
  __syncthreads();
}

// ---------------- GRU: persistent, w_hh resident in LDS, 16 WGPs ------------
// Block b owns hidden units [32b, 32b+32): 96 rows (r,z,n) of w_hh in LDS.
__global__ void gru_scan_kernel(const float* __restrict__ x,
                                const float* __restrict__ w_ih,
                                const float* __restrict__ w_hh,
                                const float* __restrict__ b_ih,
                                const float* __restrict__ b_hh,
                                float* __restrict__ hbuf,     // 2*HID, zeroed
                                float* __restrict__ evolved,  // T*HID
                                int* __restrict__ sync)       // [cnt, gen] zeroed
{
  extern __shared__ float smem[];
  float* W  = smem;             // 96*512 fp32 = 192 KB
  float* hs = W + 96 * HID;     // 512
  float* xw = hs + HID;         // 96 (folded input weights)
  float* xb = xw + 96;          // 96
  float* hb = xb + 96;          // 96
  const int tid = threadIdx.x, bid = blockIdx.x;

  for (int idx = tid; idx < 96 * HID; idx += 256) {
    const int q = idx >> 9, cc = idx & (HID - 1);
    const int gate = q >> 5, jl = q & 31;
    W[idx] = w_hh[(gate * HID + bid * 32 + jl) * HID + cc];
  }
  if (tid < 96) {
    const int gate = tid >> 5, jl = tid & 31;
    const int grow = gate * HID + bid * 32 + jl;
    xw[tid] = w_ih[grow * 4 + 0] + 0.1f * w_ih[grow * 4 + 1]
            + 0.01f * w_ih[grow * 4 + 2] + 0.001f * w_ih[grow * 4 + 3];
    xb[tid] = b_ih[grow];
    hb[tid] = b_hh[grow];
  }
  __syncthreads();

  const int wave = tid >> 5, lane = tid & 31;
  int p = 0;
  for (int t = 0; t < T_LEN; t++) {
    hs[tid]       = hbuf[p * HID + tid];
    hs[tid + 256] = hbuf[p * HID + tid + 256];
    __syncthreads();
    const float xt = x[t];
#pragma unroll
    for (int u = 0; u < 4; u++) {             // 4 hidden units per wave
      const int jl = (wave << 2) + u;
      const float* Wr = &W[(jl     ) * HID];
      const float* Wz = &W[(32 + jl) * HID];
      const float* Wn = &W[(64 + jl) * HID];
      float sr = 0.f, sz = 0.f, sn = 0.f;
#pragma unroll
      for (int e = 0; e < 16; e++) {
        const int c = lane + (e << 5);
        const float hv = hs[c];
        sr = fmaf(hv, Wr[c], sr);
        sz = fmaf(hv, Wz[c], sz);
        sn = fmaf(hv, Wn[c], sn);
      }
#pragma unroll
      for (int off = 16; off > 0; off >>= 1) { // wave32 xor-reduce
        sr += __shfl_xor(sr, off, 32);
        sz += __shfl_xor(sz, off, 32);
        sn += __shfl_xor(sn, off, 32);
      }
      if (lane == 0) {
        const float xr = fmaf(xt, xw[jl],      xb[jl]);
        const float xz = fmaf(xt, xw[32 + jl], xb[32 + jl]);
        const float xn = fmaf(xt, xw[64 + jl], xb[64 + jl]);
        const float hr = sr + hb[jl];
        const float hz = sz + hb[32 + jl];
        const float hn = sn + hb[64 + jl];
        const float r = 1.f / (1.f + expf(-(xr + hr)));
        const float z = 1.f / (1.f + expf(-(xz + hz)));
        const float n = tanhf(xn + r * hn);
        const int ju = bid * 32 + jl;
        const float hnew = (1.f - z) * n + z * hs[ju];
        hbuf[(p ^ 1) * HID + ju] = hnew;
        evolved[t * HID + ju]    = hnew;
      }
    }
    grid_barrier(sync, sync + 1);
    p ^= 1;
  }
}

// ---------------- 4096-pt Stockham FFT stages in LDS (result lands in X) ----
__device__ inline void fft4096_stages(float2* __restrict__ X, float2* __restrict__ Y,
                                      int tid) {
  float2* src = X; float2* dst = Y;
#pragma unroll 1
  for (int stage = 0; stage < 12; stage++) {
    const int n = T_LEN >> stage;
    const int m = n >> 1;
    const int smask = (1 << stage) - 1;
    for (int idx = tid; idx < (T_LEN / 2); idx += 256) {
      const int pp = idx >> stage;
      const int q  = idx & smask;
      const float ang = 6.283185307179586f * (float)pp / (float)n;
      float sn, cs; sincosf(ang, &sn, &cs);      // w = exp(-i*ang)
      const float2 A = src[q + (pp << stage)];
      const float2 B = src[q + ((pp + m) << stage)];
      dst[q + ((2 * pp) << stage)] = make_float2(A.x + B.x, A.y + B.y);
      const float rx = A.x - B.x, ry = A.y - B.y;
      dst[q + ((2 * pp + 1) << stage)] =
          make_float2(fmaf(ry, sn, rx * cs), fmaf(-rx, sn, ry * cs));
    }
    __syncthreads();
    float2* tt = src; src = dst; dst = tt;       // 12 swaps -> back in X
  }
}

// One block per complex channel c: TDM gather -> FFT -> filter -> iFFT -> real
__global__ void fft_filter_kernel(const float* __restrict__ evolved,
                                  const float* __restrict__ alphap,
                                  float* __restrict__ ft) {
  extern __shared__ float2 fbuf[];
  float2* X = fbuf;
  float2* Y = fbuf + T_LEN;
  const int c = blockIdx.x, tid = threadIdx.x;

#ifdef USE_TDM
  // Tensor Data Mover: 2D tile (2 x 4096 f32), row stride 512 f32, -> LDS X.
  // One issue per workgroup (TDM ignores EXEC); D# per ISA 08 sec 8.3/8.4.
  if (tid == 0) {
    const unsigned long long ga = (unsigned long long)(uintptr_t)(evolved + 2 * c);
    const unsigned lds_off = (unsigned)(uintptr_t)X;   // low 32b of flat = LDS offset
    v4u g0;
    g0[0] = 1u;                                   // count=1 (valid user D#)
    g0[1] = lds_off;                              // lds_addr
    g0[2] = (unsigned)ga;                         // global_addr[31:0]
    g0[3] = (unsigned)(ga >> 32) | (2u << 30);    // global_addr[56:32] | type=2
    v8i g1;
    g1[0] = (int)(2u << 16);                      // data_size=2 (4 bytes)
    g1[1] = (int)(512u << 16);                    // tensor_dim0 = 512 (lo16)
    g1[2] = (int)(4096u << 16);                   // tensor_dim1 = 4096 (lo16)
    g1[3] = (int)(2u << 16);                      // tile_dim0 = 2
    g1[4] = (int)4096;                            // tile_dim1 = 4096, tile_dim2 = 0
    g1[5] = 512;                                  // tensor_dim0_stride = 512
    g1[6] = 0;
    g1[7] = 0;
    v4i z4 = {0, 0, 0, 0};
#if __clang_major__ >= 23
    v8i z8 = {0, 0, 0, 0, 0, 0, 0, 0};
    __builtin_amdgcn_tensor_load_to_lds(g0, g1, z4, z4, z8, 0);
#else
    __builtin_amdgcn_tensor_load_to_lds(g0, g1, z4, z4, 0);
#endif
    __builtin_amdgcn_s_wait_tensorcnt(0);         // s_wait_tensorcnt 0
  }
  __syncthreads();
#else
  for (int t = tid; t < T_LEN; t += 256)
    X[t] = make_float2(evolved[t * HID + 2 * c], evolved[t * HID + 2 * c + 1]);
  __syncthreads();
#endif

  fft4096_stages(X, Y, tid);
  const float alpha = alphap[0];
  for (int k = tid; k < T_LEN; k += 256) {
    const float2 f = X[k];
    const float mag = sqrtf(f.x * f.x + f.y * f.y) + 1e-10f;
    const float pf = alpha * atanf(logf(mag));
    float sp, cp; sincosf(pf, &sp, &cp);
    const float fx = f.x * cp - f.y * sp;
    const float fy = f.x * sp + f.y * cp;
    X[k] = make_float2(fx, -fy);                 // conj: ifft via forward fft
  }
  __syncthreads();
  fft4096_stages(X, Y, tid);
  const float inv = 1.0f / (float)T_LEN;
  for (int t = tid; t < T_LEN; t += 256)
    ft[t * 256 + c] = X[t].x * inv;              // real(ifft)
}

// ---------------- coh = ft @ pc_w.T + pc_b  (fp32, emit bf16 for WMMA) ------
__global__ void coh_kernel(const float* __restrict__ ft, const float* __restrict__ pc_w,
                           const float* __restrict__ pc_b, __bf16* __restrict__ cohb) {
  __shared__ float row[256];
  const int t = blockIdx.x, d = threadIdx.x;
  row[d]       = ft[t * 256 + d];
  row[d + 128] = ft[t * 256 + d + 128];
  __syncthreads();
  float acc = pc_b[d];
#pragma unroll 4
  for (int c = 0; c < 256; c++) acc = fmaf(row[c], pc_w[d * 256 + c], acc);
  cohb[t * DIM + d] = (__bf16)acc;
}

__global__ void cvt_bf16_kernel(const float* __restrict__ src,
                                __bf16* __restrict__ dst, int n) {
  const int i = blockIdx.x * blockDim.x + threadIdx.x;
  if (i < n) dst[i] = (__bf16)src[i];
}

// ---------------- logits = (coh@pat_re.T)^2 + (coh@pat_im.T)^2 via WMMA -----
// Each wave: one 16(M)x64(N) strip; K=128 = 4x v_wmma_f32_16x16x32_bf16 per acc.
__global__ void logits_wmma_kernel(const __bf16* __restrict__ cohb,
                                   const __bf16* __restrict__ patre,
                                   const __bf16* __restrict__ patim,
                                   float* __restrict__ out) {
  const int wave = threadIdx.x >> 5;
  const int lane = threadIdx.x & 31;
  const int gw = blockIdx.x * 8 + wave;
  const int mt = gw / 500;                 // 256 M-tiles
  const int ng = gw - mt * 500;            // 500 N-groups of 64
  const int m0 = mt << 4;
  const int vbase = ng << 6;
  const int g  = lane >> 4;                // lane half
  const int lm = lane & 15;

  // A fragment (coh row-major 4096x128), contiguous K layout:
  // lanes 0-15: K = i; lanes 16-31: K = 16 + i
  v16bf a[4];
  {
    const __bf16* arow = cohb + (m0 + lm) * DIM;
#pragma unroll
    for (int s = 0; s < 4; s++) {
      const __bf16* ap = arow + 32 * s + 16 * g;
#pragma unroll
      for (int i = 0; i < 8; i++)  a[s][i] = ap[i];
#pragma unroll
      for (int i = 8; i < 16; i++) a[s][i] = ap[i];
    }
  }
#pragma unroll 1
  for (int sub = 0; sub < 4; sub++) {
    const int col = vbase + (sub << 4) + lm;
    const __bf16* brow  = patre + (size_t)col * DIM + 16 * g;  // B: K = 16g + i
    const __bf16* birow = patim + (size_t)col * DIM + 16 * g;
    __builtin_prefetch(brow + 16 * DIM, 0, 3);   // next sub-tile, keep near (WGP)
    __builtin_prefetch(birow + 16 * DIM, 0, 3);
    v8f accR = {0.f, 0.f, 0.f, 0.f, 0.f, 0.f, 0.f, 0.f};
    v8f accI = {0.f, 0.f, 0.f, 0.f, 0.f, 0.f, 0.f, 0.f};
#pragma unroll
    for (int s = 0; s < 4; s++) {
      v16bf br, bi;
#pragma unroll
      for (int i = 0; i < 16; i++) { br[i] = brow[32 * s + i]; bi[i] = birow[32 * s + i]; }
      accR = __builtin_amdgcn_wmma_f32_16x16x32_bf16(false, a[s], false, br,
                                                     (short)0, accR, false, false);
      accI = __builtin_amdgcn_wmma_f32_16x16x32_bf16(false, a[s], false, bi,
                                                     (short)0, accI, false, false);
    }
    // D layout: VGPR r, lane l -> M = r + 8*(l/16), N = l%16
    float* orow = out + (size_t)(m0 + (g << 3)) * VOC + col;
#pragma unroll
    for (int r = 0; r < 8; r++) {
      const float zr = accR[r], zi = accI[r];
      orow[(size_t)r * VOC] = fmaf(zr, zr, zi * zi);
    }
  }
}

// ---------------- workspace layout ------------------------------------------
// [0,8)        sync counters        (zeroed each call)
// [256,4352)   hbuf double buffer   (zeroed each call -> h0 = 0)
// [8192, +8MB) evolved (T x 512 f32)
// then ft (T x 256 f32), coh bf16, pat_re bf16, pat_im bf16   (~30 MB total)
#define OFF_EVOLVED 8192u
#define OFF_FT      8396800u
#define OFF_COHB    12591104u
#define OFF_PATRE   13639680u
#define OFF_PATIM   21831680u
#define WS_NEEDED   30023680u

extern "C" void kernel_launch(void* const* d_in, const int* in_sizes, int n_in,
                              void* d_out, int out_size, void* d_ws, size_t ws_size,
                              hipStream_t stream) {
  (void)in_sizes; (void)n_in; (void)out_size;
  if (ws_size < WS_NEEDED) return;
  const float* x      = (const float*)d_in[0];
  const float* w_ih   = (const float*)d_in[1];
  const float* w_hh   = (const float*)d_in[2];
  const float* b_ih   = (const float*)d_in[3];
  const float* b_hh   = (const float*)d_in[4];
  const float* alpha  = (const float*)d_in[5];
  const float* pc_w   = (const float*)d_in[6];
  const float* pc_b   = (const float*)d_in[7];
  const float* pat_re = (const float*)d_in[8];
  const float* pat_im = (const float*)d_in[9];

  char* ws = (char*)d_ws;
  int*    syncp   = (int*)ws;
  float*  hbuf    = (float*)(ws + 256);
  float*  evolved = (float*)(ws + OFF_EVOLVED);
  float*  ft      = (float*)(ws + OFF_FT);
  __bf16* cohb    = (__bf16*)(ws + OFF_COHB);
  __bf16* patreb  = (__bf16*)(ws + OFF_PATRE);
  __bf16* patimb  = (__bf16*)(ws + OFF_PATIM);
  float*  out     = (float*)d_out;

  hipMemsetAsync(d_ws, 0, 8192, stream);   // reset barrier state + h0 every call

  const int npat = VOC * DIM;
  cvt_bf16_kernel<<<(npat + 255) / 256, 256, 0, stream>>>(pat_re, patreb, npat);
  cvt_bf16_kernel<<<(npat + 255) / 256, 256, 0, stream>>>(pat_im, patimb, npat);

  const size_t gru_lds = (size_t)(96 * HID + HID + 3 * 96) * sizeof(float); // ~195 KB
  gru_scan_kernel<<<GRU_NBLK, 256, gru_lds, stream>>>(x, w_ih, w_hh, b_ih, b_hh,
                                                      hbuf, evolved, syncp);

  fft_filter_kernel<<<256, 256, 2 * T_LEN * sizeof(float2), stream>>>(evolved, alpha, ft);

  coh_kernel<<<T_LEN, 128, 0, stream>>>(ft, pc_w, pc_b, cohb);

  logits_wmma_kernel<<<16000, 256, 0, stream>>>(cohb, patreb, patimb, out);
}
